// MixtureOfRoutingAttention_15333033247310
// MI455X (gfx1250) — matrix operand order repaired and checked by reading, not compiled, not run on hardware
//
#include <hip/hip_runtime.h>
#include <cmath>

// ---------------- problem constants ----------------
#define B_   8
#define T_   512
#define D_   768
#define H_   8
#define HD_  96
#define BT_  (B_*T_)
#define SCALE_ 0.1020620726159657f   // 1/sqrt(96)

// ---------------- CDNA5 WMMA types ----------------
typedef __attribute__((ext_vector_type(16))) __bf16 v16bf;
typedef __attribute__((ext_vector_type(8)))  __bf16 bf16x8;
typedef __attribute__((ext_vector_type(4)))  __bf16 bf16x4;
typedef __attribute__((ext_vector_type(8)))  float  v8f;
typedef __attribute__((ext_vector_type(4)))  float  f4;

union FragU { v16bf v; bf16x8 h[2]; };

// A-matrix fragment (16xK=32 bf16), row-major source.
// Lanes 0-15: row m=l, K chunks {0-7, 16-23}; lanes 16-31: row m=l-16, K {8-15, 24-31}.
__device__ __forceinline__ v16bf frag_row(const __bf16* base, int stride) {
  const int l = threadIdx.x & 31, hi = l >> 4, m = l & 15;
  const __bf16* p = base + (size_t)m * stride;
  FragU f;
  f.h[0] = *(const bf16x8*)(p + hi * 8);
  f.h[1] = *(const bf16x8*)(p + 16 + hi * 8);
  return f.v;
}

// B-matrix fragment (K=32 x 16 bf16) where B[k][n] = src[n][k].
// Lane l: column n=l&15, K chunks {0-7, 16-23} (lanes 0-15) / {8-15, 24-31} (lanes 16-31) within source row n.
__device__ __forceinline__ v16bf frag_col(const __bf16* base, int stride) {
  const int l = threadIdx.x & 31, hi = l >> 4, n = l & 15;
  const __bf16* p = base + (size_t)n * stride + hi * 8;
  FragU f;
  f.h[0] = *(const bf16x8*)(p);
  f.h[1] = *(const bf16x8*)(p + 16);
  return f.v;
}

__device__ __forceinline__ v8f wmma_bf16(v16bf a, v16bf b, v8f c) {
  return __builtin_amdgcn_wmma_f32_16x16x32_bf16(false, a, false, b, (short)0, c,
                                                 false, false);
}

// ---- CDNA5 async global->LDS copy (ASYNCcnt path), per-lane 16B ----
// GV form: vdst = LDS byte address (flat addr truncates to LDS offset),
// vaddr = 64-bit global address. Tracked by ASYNCcnt.
__device__ __forceinline__ void async_ld_b128(void* lds, const void* gp) {
  asm volatile("global_load_async_to_lds_b128 %0, %1, off"
               :: "v"((unsigned int)(unsigned long long)lds), "v"(gp)
               : "memory");
}
__device__ __forceinline__ void wait_async0() {
#if __has_builtin(__builtin_amdgcn_s_wait_asynccnt)
  __builtin_amdgcn_s_wait_asynccnt(0);
#else
  asm volatile("s_wait_asynccnt 0x0" ::: "memory");
#endif
}

__device__ __forceinline__ float gelu_exact(float x) {
  return 0.5f * x * (1.0f + erff(x * 0.70710678118654752f));
}

// ======================================================================
// fp32 -> bf16 bulk converter (one-time weight pass; n multiple of 1024)
// ======================================================================
__global__ __launch_bounds__(256)
void cvt_kernel(const float* __restrict__ src, __bf16* __restrict__ dst) {
  const size_t i = ((size_t)blockIdx.x * 256 + threadIdx.x) * 4;
  f4 v = *(const f4*)(src + i);
  bf16x4 o;
  #pragma unroll
  for (int k = 0; k < 4; ++k) o[k] = (__bf16)v[k];
  *(bf16x4*)(dst + i) = o;
}

// ======================================================================
// Router: mean over T, gelu GEMV (768->128), GEMV (128->8), per-half argmax.
// Softmax is monotone, so argmax of logits == argmax of softmax.
// ======================================================================
__global__ __launch_bounds__(256)
void router_kernel(const float* __restrict__ X,
                   const float* __restrict__ w1, const float* __restrict__ b1,
                   const float* __restrict__ w2, const float* __restrict__ b2,
                   int* __restrict__ idx_s, int* __restrict__ idx_t) {
  __shared__ float xm[D_];
  __shared__ float hh[128];
  __shared__ float lg[8];
  for (int b = 0; b < B_; ++b) {
    for (int d = threadIdx.x; d < D_; d += 256) {
      float s = 0.f;
      const float* xp = X + ((size_t)b * T_) * D_ + d;
      for (int t = 0; t < T_; ++t) s += xp[(size_t)t * D_];
      xm[d] = s * (1.0f / T_);
    }
    __syncthreads();
    if (threadIdx.x < 128) {
      float s = b1[threadIdx.x];
      const float* wr = w1 + (size_t)threadIdx.x * D_;
      for (int d = 0; d < D_; ++d) s += xm[d] * wr[d];
      hh[threadIdx.x] = gelu_exact(s);
    }
    __syncthreads();
    if (threadIdx.x < 8) {
      float s = b2[threadIdx.x];
      const float* wr = w2 + (size_t)threadIdx.x * 128;
      for (int d = 0; d < 128; ++d) s += hh[d] * wr[d];
      lg[threadIdx.x] = s;
    }
    __syncthreads();
    if (threadIdx.x == 0) {
      int is = 0, it = 0;
      for (int i = 1; i < 4; ++i) { if (lg[i]     > lg[is])     is = i; }
      for (int i = 1; i < 4; ++i) { if (lg[4 + i] > lg[4 + it]) it = i; }
      idx_s[b] = is; idx_t[b] = it;
    }
    __syncthreads();
  }
}

// ======================================================================
// LayerNorm over D=768, writes optional f32 and/or bf16 outputs.
// ======================================================================
__global__ __launch_bounds__(256)
void layernorm_kernel(const float* __restrict__ X,
                      const float* __restrict__ g, const float* __restrict__ be,
                      float* __restrict__ outf, __bf16* __restrict__ outb) {
  __shared__ float red[256];
  const int row = blockIdx.x;
  const float* xr = X + (size_t)row * D_;
  float s = 0.f, s2 = 0.f;
  for (int d = threadIdx.x; d < D_; d += 256) { float v = xr[d]; s += v; s2 += v * v; }
  red[threadIdx.x] = s; __syncthreads();
  for (int o = 128; o > 0; o >>= 1) { if (threadIdx.x < o) red[threadIdx.x] += red[threadIdx.x + o]; __syncthreads(); }
  const float mean = red[0] * (1.0f / D_);
  __syncthreads();
  red[threadIdx.x] = s2; __syncthreads();
  for (int o = 128; o > 0; o >>= 1) { if (threadIdx.x < o) red[threadIdx.x] += red[threadIdx.x + o]; __syncthreads(); }
  const float var = red[0] * (1.0f / D_) - mean * mean;
  const float inv = rsqrtf(var + 1e-5f);
  for (int d = threadIdx.x; d < D_; d += 256) {
    const float v = (xr[d] - mean) * inv * g[d] + be[d];
    if (outf) outf[(size_t)row * D_ + d] = v;
    if (outb) outb[(size_t)row * D_ + d] = (__bf16)v;
  }
}

// ======================================================================
// Batched-expert bf16 GEMM:  out[b, m, n] = A[b, m, :] . W[e(b), n, :] + bias
// A, W bf16 (W pre-converted); both tiles staged via async global->LDS.
// Epilogue: +bias, optional GELU, optional fp32 residual, fp32/bf16 stores.
// Tile: 128(M) x 64(N) x 32(K); 8 waves, each owning a 32x32 patch (2x2 WMMA).
// ======================================================================
#define GBM 128
#define GBN 64
#define GBK 32
#define GLD (GBK + 8)

__global__ __launch_bounds__(256)
void gemm_bf16_kernel(const __bf16* __restrict__ A,
                      const __bf16* __restrict__ W,
                      const float* __restrict__ bias,
                      const int* __restrict__ eidx, long wstride, long bstride,
                      int M, int N, int Kd,
                      const float* __restrict__ res,
                      float* __restrict__ outf,
                      __bf16* __restrict__ outb,
                      int do_gelu) {
  __shared__ __bf16 As[GBM][GLD];
  __shared__ __bf16 Ws[GBN][GLD];

  const int b   = blockIdx.z;
  const int bm0 = blockIdx.y * GBM;
  const int bn0 = blockIdx.x * GBN;
  const int tid = threadIdx.x;
  const int e   = eidx ? eidx[b] : 0;
  const __bf16* Wb = W + (long)e * wstride;
  const float*  Bb = bias + (long)e * bstride;
  const __bf16* Ab = A + ((size_t)b * M) * Kd;

  const int wave = tid >> 5;
  const int wm = wave >> 1;   // 0..3 -> M sub-tile (32 rows)
  const int wn = wave & 1;    // 0..1 -> N sub-tile (32 cols)

  const v8f zero = {0.f, 0.f, 0.f, 0.f, 0.f, 0.f, 0.f, 0.f};
  v8f acc[2][2];
  #pragma unroll
  for (int i = 0; i < 2; ++i)
    #pragma unroll
    for (int j = 0; j < 2; ++j) acc[i][j] = zero;

  const int arow = tid >> 1, acseg = (tid & 1) * 16;   // A: 128 rows x 2 x 16 bf16
  const int wrow = tid >> 1, wseg = (tid & 1) * 16;    // W: 64 rows x 2 x 16 bf16 (tid<128)

  for (int k0 = 0; k0 < Kd; k0 += GBK) {
    __syncthreads();   // protect previous iteration's fragment reads
    { // A tile: two async 16B copies per lane
      const __bf16* srcA = Ab + (size_t)(bm0 + arow) * Kd + k0 + acseg;
      async_ld_b128(&As[arow][acseg],     srcA);
      async_ld_b128(&As[arow][acseg + 8], srcA + 8);
    }
    if (tid < 128) { // W tile: one async 16B copy per lane
      const __bf16* srcW = Wb + (size_t)(bn0 + wrow) * Kd + k0 + wseg;
      async_ld_b128(&Ws[wrow][wseg], srcW);
      if (k0 + GBK < Kd)
        __builtin_prefetch(srcW + GBK, 0, 1);   // global_prefetch_b8
    }
    wait_async0();
    __syncthreads();

    const v16bf a0 = frag_row(&As[wm * 32][0],      GLD);
    const v16bf a1 = frag_row(&As[wm * 32 + 16][0], GLD);
    const v16bf b0 = frag_col(&Ws[wn * 32][0],      GLD);
    const v16bf b1 = frag_col(&Ws[wn * 32 + 16][0], GLD);
    acc[0][0] = wmma_bf16(a0, b0, acc[0][0]);
    acc[0][1] = wmma_bf16(a0, b1, acc[0][1]);
    acc[1][0] = wmma_bf16(a1, b0, acc[1][0]);
    acc[1][1] = wmma_bf16(a1, b1, acc[1][1]);
  }

  // epilogue; C layout: VGPR r -> M = 8*(lane>>4)+r, N = lane&15
  const int l = tid & 31, hi = l >> 4, n = l & 15;
  #pragma unroll
  for (int i = 0; i < 2; ++i)
    #pragma unroll
    for (int j = 0; j < 2; ++j) {
      const int gn = bn0 + wn * 32 + j * 16 + n;
      const float bv = Bb[gn];
      #pragma unroll
      for (int r = 0; r < 8; ++r) {
        const int gm = bm0 + wm * 32 + i * 16 + hi * 8 + r;
        float v = acc[i][j][r] + bv;
        if (do_gelu) v = gelu_exact(v);
        const size_t off = ((size_t)b * M + gm) * N + gn;
        if (res)  v += res[off];
        if (outf) outf[off] = v;
        if (outb) outb[off] = (__bf16)v;
      }
    }
}

// ======================================================================
// Flash attention, bf16 in / bf16 out, head dim 96.
// Grid (T/128, H, B), 256 threads. Wave w owns q rows [blk*128+16w, +16).
// QK^T: 6 WMMAs (3 d-chunks x 2 kv halves); online softmax via width-16 shfl_xor reductions;
// P restaged via per-wave LDS into A-frag layout; PV: 6 WMMAs per kv tile.
// K tile staged with async global->LDS; V transposed through VGPRs.
// ======================================================================
#define QB 128
__global__ __launch_bounds__(256)
void attn_kernel(const __bf16* __restrict__ Q, int qstride,
                 const __bf16* __restrict__ Kp, int kstride,
                 const __bf16* __restrict__ Vp, int vstride,
                 __bf16* __restrict__ O, int causal) {
  __shared__ __bf16 Ks[32][HD_ + 8];       // kv rows x head dim
  __shared__ __bf16 Vt[HD_][32 + 8];       // transposed V tile (d major)
  __shared__ __bf16 Pl[8][16][32 + 8];     // per-wave P staging

  const int b = blockIdx.z, h = blockIdx.y, qblk = blockIdx.x;
  const int tid = threadIdx.x;
  const int wave = tid >> 5;
  const int l = tid & 31, hi = l >> 4, n = l & 15;
  const int q0 = qblk * QB + wave * 16;

  v16bf qf[3];
  const __bf16* qbase = Q + ((size_t)b * T_ + q0) * qstride + h * HD_;
  #pragma unroll
  for (int c = 0; c < 3; ++c) qf[c] = frag_row(qbase + c * 32, qstride);

  const v8f zero = {0.f, 0.f, 0.f, 0.f, 0.f, 0.f, 0.f, 0.f};
  v8f po[6];
  #pragma unroll
  for (int t = 0; t < 6; ++t) po[t] = zero;
  float mrun[8], lrun[8];
  #pragma unroll
  for (int r = 0; r < 8; ++r) { mrun[r] = -1e30f; lrun[r] = 0.0f; }

  const int nkv = causal ? (qblk + 1) * 4 : (T_ / 32);
  for (int jt = 0; jt < nkv; ++jt) {
    const int kv0 = jt * 32;
    __syncthreads();
    if (tid < 192) {                         // 32 rows x 6 chunks of 16
      const int row = tid / 6, ch = tid % 6;
      const __bf16* ksrc = Kp + ((size_t)b * T_ + kv0 + row) * kstride + h * HD_ + ch * 16;
      async_ld_b128(&Ks[row][ch * 16],     ksrc);      // ASYNCcnt path
      async_ld_b128(&Ks[row][ch * 16 + 8], ksrc + 8);
      const __bf16* vsrc = Vp + ((size_t)b * T_ + kv0 + row) * vstride + h * HD_ + ch * 16;
      bf16x8 v0 = *(const bf16x8*)(vsrc);
      bf16x8 v1 = *(const bf16x8*)(vsrc + 8);
      #pragma unroll
      for (int e2 = 0; e2 < 8; ++e2) {
        Vt[ch * 16 + e2][row]     = v0[e2];
        Vt[ch * 16 + 8 + e2][row] = v1[e2];
      }
    }
    wait_async0();
    __syncthreads();

    v8f s[2]; s[0] = zero; s[1] = zero;
    #pragma unroll
    for (int c = 0; c < 3; ++c) {
      #pragma unroll
      for (int jn = 0; jn < 2; ++jn) {
        const v16bf kb = frag_col(&Ks[jn * 16][c * 32], HD_ + 8);
        s[jn] = wmma_bf16(qf[c], kb, s[jn]);
      }
    }

    float alpha[8];
    #pragma unroll
    for (int r = 0; r < 8; ++r) {
      float s0 = s[0][r] * SCALE_;
      float s1 = s[1][r] * SCALE_;
      if (causal) {
        const int gq = q0 + hi * 8 + r;
        if (kv0 + n      > gq) s0 = -1e30f;
        if (kv0 + 16 + n > gq) s1 = -1e30f;
      }
      float mx = fmaxf(s0, s1);
      #pragma unroll
      for (int off = 1; off < 16; off <<= 1) mx = fmaxf(mx, __shfl_xor(mx, off, 16));
      const float mnew = fmaxf(mrun[r], mx);
      const float a  = __expf(mrun[r] - mnew);
      const float p0 = __expf(s0 - mnew);
      const float p1 = __expf(s1 - mnew);
      float ps = p0 + p1;
      #pragma unroll
      for (int off = 1; off < 16; off <<= 1) ps += __shfl_xor(ps, off, 16);
      lrun[r] = lrun[r] * a + ps;
      mrun[r] = mnew;
      alpha[r] = a;
      Pl[wave][hi * 8 + r][n]      = (__bf16)p0;   // C-layout -> row-major LDS
      Pl[wave][hi * 8 + r][16 + n] = (__bf16)p1;
    }
    #pragma unroll
    for (int t = 0; t < 6; ++t)
      #pragma unroll
      for (int r = 0; r < 8; ++r) po[t][r] *= alpha[r];

    const v16bf pf = frag_row(&Pl[wave][0][0], 32 + 8);  // wave-local, DS in-order
    #pragma unroll
    for (int t = 0; t < 6; ++t) {
      const v16bf vb = frag_col(&Vt[t * 16][0], 32 + 8);
      po[t] = wmma_bf16(pf, vb, po[t]);
    }
  }

  #pragma unroll
  for (int t = 0; t < 6; ++t)
    #pragma unroll
    for (int r = 0; r < 8; ++r) {
      const float v = po[t][r] / lrun[r];
      O[((size_t)b * T_ + q0 + hi * 8 + r) * D_ + h * HD_ + t * 16 + n] = (__bf16)v;
    }
}

// ======================================================================
// Host-side orchestration
// ======================================================================
extern "C" void kernel_launch(void* const* d_in, const int* in_sizes, int n_in,
                              void* d_out, int out_size, void* d_ws, size_t ws_size,
                              hipStream_t stream) {
  (void)in_sizes; (void)n_in; (void)out_size; (void)ws_size;
  const float* x      = (const float*)d_in[0];
  const float* rw1    = (const float*)d_in[1];
  const float* rb1    = (const float*)d_in[2];
  const float* rw2    = (const float*)d_in[3];
  const float* rb2    = (const float*)d_in[4];
  const float* ns_g   = (const float*)d_in[5];
  const float* ns_b   = (const float*)d_in[6];
  const float* nt_g   = (const float*)d_in[7];
  const float* nt_b   = (const float*)d_in[8];
  const float* nm_g   = (const float*)d_in[9];
  const float* nm_b   = (const float*)d_in[10];
  const float* sp_wqkv = (const float*)d_in[11];
  const float* sp_bqkv = (const float*)d_in[12];
  const float* sp_wo   = (const float*)d_in[13];
  const float* sp_bo   = (const float*)d_in[14];
  const float* tp_wq   = (const float*)d_in[15];
  const float* tp_bq   = (const float*)d_in[16];
  const float* tp_wk   = (const float*)d_in[17];
  const float* tp_bk   = (const float*)d_in[18];
  const float* tp_wv   = (const float*)d_in[19];
  const float* tp_bv   = (const float*)d_in[20];
  const float* tp_wo   = (const float*)d_in[21];
  const float* tp_bo   = (const float*)d_in[22];
  const float* c_wqkv  = (const float*)d_in[23];
  const float* c_bqkv  = (const float*)d_in[24];
  const float* c_wo    = (const float*)d_in[25];
  const float* c_bo    = (const float*)d_in[26];
  const float* m_w1    = (const float*)d_in[27];
  const float* m_b1    = (const float*)d_in[28];
  const float* m_w2    = (const float*)d_in[29];
  const float* m_b2    = (const float*)d_in[30];

  // workspace carving (liveness-based reuse)
  char* p = (char*)d_ws;
  auto alloc = [&](size_t bytes) -> char* {
    char* r = p; p += (bytes + 255) & ~(size_t)255; return r;
  };
  const size_t ND = (size_t)BT_ * D_;
  int*    idx_s  = (int*)alloc(2 * B_ * sizeof(int));
  int*    idx_t  = idx_s + B_;
  __bf16* bufA   = (__bf16*)alloc(ND * 3 * 2);  // spatial qkv -> cq/ck/cv
  __bf16* bufB   = (__bf16*)alloc(ND * 2);      // xn_s -> attn_s -> fused
  __bf16* bufC   = (__bf16*)alloc(ND * 2);      // xt_bf -> attn_t -> xm
  __bf16* qt     = (__bf16*)alloc(ND * 2);
  __bf16* kt     = (__bf16*)alloc(ND * 2);
  __bf16* vt     = (__bf16*)alloc(ND * 2);
  __bf16* spo    = (__bf16*)alloc(ND * 2);      // spatial_out
  __bf16* tpo    = (__bf16*)alloc(ND * 2);      // temporal_out
  __bf16* ybuf   = (__bf16*)alloc(ND * 4 * 2);  // mlp hidden
  float*  f32buf = (float*)alloc(ND * 4);       // xt_f32 -> x1

  // one-time fp32 -> bf16 weight conversion (halves steady-state W traffic,
  // keeps the GEMM hot loop conversion-free, enables async W staging)
  auto cvt = [&](const float* src, size_t nelem) -> __bf16* {
    __bf16* dst = (__bf16*)alloc(nelem * 2);
    cvt_kernel<<<dim3((unsigned)(nelem / 1024)), 256, 0, stream>>>(src, dst);
    return dst;
  };
  const long DD = (long)D_ * D_;
  __bf16* w_spqkv = cvt(sp_wqkv, (size_t)4 * 3 * D_ * D_);
  __bf16* w_spo   = cvt(sp_wo,   (size_t)4 * D_ * D_);
  __bf16* w_tq    = cvt(tp_wq,   (size_t)4 * D_ * D_);
  __bf16* w_tk    = cvt(tp_wk,   (size_t)4 * D_ * D_);
  __bf16* w_tv    = cvt(tp_wv,   (size_t)4 * D_ * D_);
  __bf16* w_to    = cvt(tp_wo,   (size_t)4 * D_ * D_);
  __bf16* w_cqkv  = cvt(c_wqkv,  (size_t)3 * D_ * D_);
  __bf16* w_co    = cvt(c_wo,    (size_t)D_ * D_);
  __bf16* w_m1    = cvt(m_w1,    (size_t)4 * D_ * D_);
  __bf16* w_m2    = cvt(m_w2,    (size_t)4 * D_ * D_);

  const dim3 blk(256);
  const dim3 gLN(BT_);

  // router + layernorms of x
  router_kernel<<<1, blk, 0, stream>>>(x, rw1, rb1, rw2, rb2, idx_s, idx_t);
  layernorm_kernel<<<gLN, blk, 0, stream>>>(x, ns_g, ns_b, nullptr, bufB);  // xn_s
  layernorm_kernel<<<gLN, blk, 0, stream>>>(x, nt_g, nt_b, f32buf, bufC);   // xt

  // spatial branch: routed qkv -> attention -> routed out-proj
  gemm_bf16_kernel<<<dim3(3 * D_ / GBN, T_ / GBM, B_), blk, 0, stream>>>(
      bufB, w_spqkv, sp_bqkv, idx_s, 3 * DD, 3L * D_, T_, 3 * D_, D_,
      nullptr, nullptr, bufA, 0);
  // temporal branch projections (routed)
  gemm_bf16_kernel<<<dim3(D_ / GBN, T_ / GBM, B_), blk, 0, stream>>>(
      bufC, w_tq, tp_bq, idx_t, DD, (long)D_, T_, D_, D_, nullptr, nullptr, qt, 0);
  gemm_bf16_kernel<<<dim3(D_ / GBN, T_ / GBM, B_), blk, 0, stream>>>(
      bufC, w_tk, tp_bk, idx_t, DD, (long)D_, T_, D_, D_, nullptr, nullptr, kt, 0);
  gemm_bf16_kernel<<<dim3(D_ / GBN, T_ / GBM, B_), blk, 0, stream>>>(
      bufC, w_tv, tp_bv, idx_t, DD, (long)D_, T_, D_, D_, nullptr, nullptr, vt, 0);

  const dim3 gATT(T_ / QB, H_, B_);
  // spatial self-attention (no mask)
  attn_kernel<<<gATT, blk, 0, stream>>>(bufA, 3 * D_, bufA + D_, 3 * D_,
                                        bufA + 2 * D_, 3 * D_, bufB, 0);
  gemm_bf16_kernel<<<dim3(D_ / GBN, T_ / GBM, B_), blk, 0, stream>>>(
      bufB, w_spo, sp_bo, idx_s, DD, (long)D_, T_, D_, D_, nullptr, nullptr, spo, 0);

  // temporal causal attention + out-proj with residual xt
  attn_kernel<<<gATT, blk, 0, stream>>>(qt, D_, kt, D_, vt, D_, bufC, 1);
  gemm_bf16_kernel<<<dim3(D_ / GBN, T_ / GBM, B_), blk, 0, stream>>>(
      bufC, w_to, tp_bo, idx_t, DD, (long)D_, T_, D_, D_, f32buf, nullptr, tpo, 0);

  // cross projections (shared weights): q from spatial_out, k/v from temporal_out
  gemm_bf16_kernel<<<dim3(D_ / GBN, T_ / GBM, B_), blk, 0, stream>>>(
      spo, w_cqkv, c_bqkv, nullptr, 0, 0, T_, D_, D_, nullptr, nullptr, bufA, 0);
  gemm_bf16_kernel<<<dim3(D_ / GBN, T_ / GBM, B_), blk, 0, stream>>>(
      tpo, w_cqkv + DD, c_bqkv + D_, nullptr, 0, 0, T_, D_, D_,
      nullptr, nullptr, bufA + ND, 0);
  gemm_bf16_kernel<<<dim3(D_ / GBN, T_ / GBM, B_), blk, 0, stream>>>(
      tpo, w_cqkv + 2 * DD, c_bqkv + 2 * D_, nullptr, 0, 0, T_, D_, D_,
      nullptr, nullptr, bufA + 2 * ND, 0);

  // cross attention + out-proj with residual x  -> x1 (f32)
  attn_kernel<<<gATT, blk, 0, stream>>>(bufA, D_, bufA + ND, D_, bufA + 2 * ND, D_,
                                        bufB, 0);
  gemm_bf16_kernel<<<dim3(D_ / GBN, T_ / GBM, B_), blk, 0, stream>>>(
      bufB, w_co, c_bo, nullptr, 0, 0, T_, D_, D_, x, f32buf, nullptr, 0);

  // MLP: LN -> gelu(W1) -> W2 + residual x1 -> d_out (f32)
  layernorm_kernel<<<gLN, blk, 0, stream>>>(f32buf, nm_g, nm_b, nullptr, bufC);
  gemm_bf16_kernel<<<dim3(4 * D_ / GBN, T_ / GBM, B_), blk, 0, stream>>>(
      bufC, w_m1, m_b1, nullptr, 0, 0, T_, 4 * D_, D_, nullptr, nullptr, ybuf, 1);
  gemm_bf16_kernel<<<dim3(D_ / GBN, T_ / GBM, B_), blk, 0, stream>>>(
      ybuf, w_m2, m_b2, nullptr, 0, 0, T_, D_, 4 * D_, f32buf, (float*)d_out,
      nullptr, 0);
}
